// MusicianLayer_59115929862667
// MI455X (gfx1250) — matrix operand order repaired and checked by reading, not compiled, not run on hardware
//
#include <hip/hip_runtime.h>

// ---------------------------------------------------------------------------
// MI455X / gfx1250 implementation.
// All matmuls via v_wmma_f32_16x16x32_bf16 (wave32, 16x16x32 fragments).
// Fragment LDS traffic is fully vectorized into ds_load_b128 pairs.
// ---------------------------------------------------------------------------

typedef __attribute__((ext_vector_type(16))) __bf16        v16bf;
typedef __attribute__((ext_vector_type(8)))  float         v8f;
typedef __attribute__((ext_vector_type(4)))  unsigned int  u32x4;

union AFrag { v16bf v; unsigned short u[16]; u32x4 q[2]; };
union CFrag { v8f   v; float          f[8];  };

#define NB   4
#define NT   2048
#define ND   1024
#define NH   16
#define NDH  64
#define NE   8
#define NDE  256
#define MTOK (NB * NT)        // 8192 tokens
#define NQKV (3 * ND)         // 3072
#define NGU  (NE * NDE)       // 2048

__device__ __forceinline__ unsigned short f2bf(float f) {
  union { float f; unsigned int u; } c; c.f = f;
  unsigned int u = c.u;
  u += 0x7FFFu + ((u >> 16) & 1u);       // round-to-nearest-even
  return (unsigned short)(u >> 16);
}

// ---------------------------------------------------------------------------
// Tiled GEMM:  C[M,N] = A[M,K](bf16) * W[N,K]^T(bf16)  (+bias) (+residual)
// Block 256 threads = 8 waves (2x4), tile 128x128x64, per-wave 64x32.
// A-fragment per lane = two contiguous 16B chunks; B-fragment = one 32B run.
// ---------------------------------------------------------------------------
#define TM 128
#define TN 128
#define TK 64
#define LDSS 72   // padded row stride (bf16 elems); 144B keeps 16B alignment

template <bool OUT_BF16>
__global__ __launch_bounds__(256) void gemm_bf16_kernel(
    const unsigned short* __restrict__ A,   // [M,K]
    const unsigned short* __restrict__ W,   // [N,K]
    const float* __restrict__ bias,         // [N] or nullptr
    const float* __restrict__ res,          // [M,N] or nullptr
    void* __restrict__ Cout,                // f32 or bf16 [M,N]
    int M, int N, int K) {
  __shared__ unsigned short As[TM * LDSS];
  __shared__ unsigned short Bs[TN * LDSS];

  const int tid  = threadIdx.x;
  const int lane = tid & 31;
  const int wave = tid >> 5;
  const int wm   = wave >> 2;      // 0..1
  const int wn   = wave & 3;       // 0..3
  const int half = lane >> 4;
  const int lr   = lane & 15;

  const int blockN = blockIdx.x * TN;
  const int blockM = blockIdx.y * TM;

  CFrag acc[4][2];
#pragma unroll
  for (int i = 0; i < 4; ++i)
#pragma unroll
    for (int j = 0; j < 2; ++j)
#pragma unroll
      for (int r = 0; r < 8; ++r) acc[i][j].f[r] = 0.f;

  const int nKT = K / TK;
  for (int kt = 0; kt < nKT; ++kt) {
    const int kbase = kt * TK;
    // Stage 128x64 A and 128x64 W tiles (each thread: 4x 16B vectors per tile)
#pragma unroll
    for (int v = 0; v < 4; ++v) {
      int idx = v * 256 + tid;          // 0..1023
      int row = idx >> 3;
      int cv  = idx & 7;
      *(u32x4*)(&As[row * LDSS + cv * 8]) =
          *(const u32x4*)(A + (size_t)(blockM + row) * K + kbase + cv * 8);
      *(u32x4*)(&Bs[row * LDSS + cv * 8]) =
          *(const u32x4*)(W + (size_t)(blockN + row) * K + kbase + cv * 8);
    }
    // Prefetch next k-tile (lowers to global_prefetch_b8)
    if (kt + 1 < nKT) {
      int row = tid >> 1, cv = tid & 1;
      __builtin_prefetch(A + (size_t)(blockM + row) * K + kbase + TK + cv * 32, 0, 3);
      __builtin_prefetch(W + (size_t)(blockN + row) * K + kbase + TK + cv * 32, 0, 3);
    }
    __syncthreads();

#pragma unroll
    for (int kk = 0; kk < TK; kk += 32) {
      AFrag b[2];
#pragma unroll
      for (int fn = 0; fn < 2; ++fn) {
        int n0 = wn * 32 + fn * 16;
        const u32x4* bp = (const u32x4*)(&Bs[(n0 + lr) * LDSS + kk + half * 16]);
        b[fn].q[0] = bp[0];
        b[fn].q[1] = bp[1];
      }
#pragma unroll
      for (int fm = 0; fm < 4; ++fm) {
        int m0 = wm * 64 + fm * 16;
        AFrag a;
        a.q[0] = *(const u32x4*)(&As[(m0 + lr) * LDSS + kk + half * 16]);
        a.q[1] = *(const u32x4*)(&As[(m0 + lr) * LDSS + kk + half * 16 + 8]);
#pragma unroll
        for (int fn = 0; fn < 2; ++fn)
          acc[fm][fn].v = __builtin_amdgcn_wmma_f32_16x16x32_bf16(
              false, a.v, false, b[fn].v, (short)0, acc[fm][fn].v, false, false);
      }
    }
    __syncthreads();
  }

#pragma unroll
  for (int fm = 0; fm < 4; ++fm)
#pragma unroll
    for (int fn = 0; fn < 2; ++fn) {
      int col = blockN + wn * 32 + fn * 16 + lr;
      float bv = bias ? bias[col] : 0.f;
#pragma unroll
      for (int r = 0; r < 8; ++r) {
        int row = blockM + wm * 64 + fm * 16 + half * 8 + r;
        float val = acc[fm][fn].f[r] + bv;
        if (res) val += res[(size_t)row * N + col];
        if (OUT_BF16)
          ((unsigned short*)Cout)[(size_t)row * N + col] = f2bf(val);
        else
          ((float*)Cout)[(size_t)row * N + col] = val;
      }
    }
}

// ---------------------------------------------------------------------------
// Fused gate/up GEMM + SiLU(hg)*hu*router_weight epilogue -> bf16 act2.
// ---------------------------------------------------------------------------
__global__ __launch_bounds__(256) void gemm_gateup_kernel(
    const unsigned short* __restrict__ A,    // [M,K]  h2 bf16
    const unsigned short* __restrict__ Wgm,  // [NGU,K]
    const unsigned short* __restrict__ Wum,  // [NGU,K]
    const float* __restrict__ wgt,           // [M,NE] router weights
    unsigned short* __restrict__ act2,       // [M,NGU]
    int M, int N, int K) {
  __shared__ unsigned short As[TM * LDSS];
  __shared__ unsigned short Bg[TN * LDSS];
  __shared__ unsigned short Bu[TN * LDSS];

  const int tid  = threadIdx.x;
  const int lane = tid & 31;
  const int wave = tid >> 5;
  const int wm   = wave >> 2;
  const int wn   = wave & 3;
  const int half = lane >> 4;
  const int lr   = lane & 15;

  const int blockN = blockIdx.x * TN;
  const int blockM = blockIdx.y * TM;

  CFrag ag[4][2], au[4][2];
#pragma unroll
  for (int i = 0; i < 4; ++i)
#pragma unroll
    for (int j = 0; j < 2; ++j)
#pragma unroll
      for (int r = 0; r < 8; ++r) { ag[i][j].f[r] = 0.f; au[i][j].f[r] = 0.f; }

  const int nKT = K / TK;
  for (int kt = 0; kt < nKT; ++kt) {
    const int kbase = kt * TK;
#pragma unroll
    for (int v = 0; v < 4; ++v) {
      int idx = v * 256 + tid;
      int row = idx >> 3;
      int cv  = idx & 7;
      *(u32x4*)(&As[row * LDSS + cv * 8]) =
          *(const u32x4*)(A + (size_t)(blockM + row) * K + kbase + cv * 8);
      *(u32x4*)(&Bg[row * LDSS + cv * 8]) =
          *(const u32x4*)(Wgm + (size_t)(blockN + row) * K + kbase + cv * 8);
      *(u32x4*)(&Bu[row * LDSS + cv * 8]) =
          *(const u32x4*)(Wum + (size_t)(blockN + row) * K + kbase + cv * 8);
    }
    __syncthreads();

#pragma unroll
    for (int kk = 0; kk < TK; kk += 32) {
      AFrag bg[2], bu[2];
#pragma unroll
      for (int fn = 0; fn < 2; ++fn) {
        int n0 = wn * 32 + fn * 16;
        const u32x4* gp = (const u32x4*)(&Bg[(n0 + lr) * LDSS + kk + half * 16]);
        const u32x4* up = (const u32x4*)(&Bu[(n0 + lr) * LDSS + kk + half * 16]);
        bg[fn].q[0] = gp[0]; bg[fn].q[1] = gp[1];
        bu[fn].q[0] = up[0]; bu[fn].q[1] = up[1];
      }
#pragma unroll
      for (int fm = 0; fm < 4; ++fm) {
        int m0 = wm * 64 + fm * 16;
        AFrag a;
        a.q[0] = *(const u32x4*)(&As[(m0 + lr) * LDSS + kk + half * 16]);
        a.q[1] = *(const u32x4*)(&As[(m0 + lr) * LDSS + kk + half * 16 + 8]);
#pragma unroll
        for (int fn = 0; fn < 2; ++fn) {
          ag[fm][fn].v = __builtin_amdgcn_wmma_f32_16x16x32_bf16(
              false, a.v, false, bg[fn].v, (short)0, ag[fm][fn].v, false, false);
          au[fm][fn].v = __builtin_amdgcn_wmma_f32_16x16x32_bf16(
              false, a.v, false, bu[fn].v, (short)0, au[fm][fn].v, false, false);
        }
      }
    }
    __syncthreads();
  }

#pragma unroll
  for (int fm = 0; fm < 4; ++fm)
#pragma unroll
    for (int fn = 0; fn < 2; ++fn) {
      int col = blockN + wn * 32 + fn * 16 + lr;
      int e   = col >> 8;                          // col / NDE
#pragma unroll
      for (int r = 0; r < 8; ++r) {
        int row = blockM + wm * 64 + fm * 16 + half * 8 + r;
        float hg = ag[fm][fn].f[r];
        float hu = au[fm][fn].f[r];
        float silu = hg / (1.f + __expf(-hg));
        float wv = wgt[(size_t)row * NE + e];
        act2[(size_t)row * NGU + col] = f2bf(silu * hu * wv);
      }
    }
}

// ---------------------------------------------------------------------------
// Flash attention: one block = 64 query rows of one (b,h); 4 waves x 16 rows.
// S = Q K^T / 8 via WMMA, online softmax, O += P V via WMMA (P via LDS).
// V is staged TRANSPOSED (Vt[d][k]) so P*V B-fragments are ds_load_b128 pairs.
// ---------------------------------------------------------------------------
#define ASTR 72   // padded row stride for 64-wide tiles (144B, 16B-aligned)

__global__ __launch_bounds__(128) void attn_kernel(
    const unsigned short* __restrict__ qkv,   // [B*T, 3D] bf16
    unsigned short* __restrict__ O) {         // [B*T, D]  bf16
  const int bh = blockIdx.y;
  const int b  = bh >> 4;
  const int h  = bh & 15;
  const int qb = blockIdx.x * 64;

  __shared__ unsigned short Qs[64 * ASTR];
  __shared__ unsigned short Ks[64 * ASTR];
  __shared__ unsigned short Vt[64 * ASTR];   // transposed: Vt[d][k]
  __shared__ unsigned short Ps[4 * 16 * ASTR];

  const int tid  = threadIdx.x;
  const int lane = tid & 31;
  const int wave = tid >> 5;
  const int half = lane >> 4;
  const int lr   = lane & 15;

  const size_t rs = (size_t)(3 * ND);
  const unsigned short* qp = qkv + (size_t)b * NT * rs + (size_t)h * NDH;
  const unsigned short* kp = qp + ND;
  const unsigned short* vp = qp + 2 * ND;

  // Load Q tile (64x64): 512 16B vectors over 128 threads
#pragma unroll
  for (int v = 0; v < 4; ++v) {
    int idx = v * 128 + tid;
    int row = idx >> 3;
    int cv  = idx & 7;
    *(u32x4*)(&Qs[row * ASTR + cv * 8]) =
        *(const u32x4*)(qp + (size_t)(qb + row) * rs + cv * 8);
  }

  float mst[8], lst[8];
  CFrag o[4];
#pragma unroll
  for (int r = 0; r < 8; ++r) { mst[r] = -3.0e38f; lst[r] = 0.f; }
#pragma unroll
  for (int c = 0; c < 4; ++c)
#pragma unroll
    for (int r = 0; r < 8; ++r) o[c].f[r] = 0.f;

  for (int kb = 0; kb < NT / 64; ++kb) {
    __syncthreads();   // prev K/V consumed (and Q ready on first iter)
#pragma unroll
    for (int v = 0; v < 4; ++v) {
      int idx = v * 128 + tid;
      int row = idx >> 3;         // key index within tile
      int cv  = idx & 7;
      *(u32x4*)(&Ks[row * ASTR + cv * 8]) =
          *(const u32x4*)(kp + (size_t)(kb * 64 + row) * rs + cv * 8);
      // V: load 8 dims of one key, scatter into transposed tile
      union { u32x4 q; unsigned short u[8]; } vv;
      vv.q = *(const u32x4*)(vp + (size_t)(kb * 64 + row) * rs + cv * 8);
#pragma unroll
      for (int e = 0; e < 8; ++e)
        Vt[(cv * 8 + e) * ASTR + row] = vv.u[e];
    }
    __syncthreads();

    // S = Q K^T (16 q-rows x 64 keys per wave)
    CFrag s[4];
#pragma unroll
    for (int fn = 0; fn < 4; ++fn)
#pragma unroll
      for (int r = 0; r < 8; ++r) s[fn].f[r] = 0.f;
#pragma unroll
    for (int kk = 0; kk < 64; kk += 32) {
      AFrag a;
      a.q[0] = *(const u32x4*)(&Qs[(wave * 16 + lr) * ASTR + kk + half * 16]);
      a.q[1] = *(const u32x4*)(&Qs[(wave * 16 + lr) * ASTR + kk + half * 16 + 8]);
#pragma unroll
      for (int fn = 0; fn < 4; ++fn) {
        AFrag bb;
        const u32x4* bp = (const u32x4*)(&Ks[(fn * 16 + lr) * ASTR + kk + half * 16]);
        bb.q[0] = bp[0];
        bb.q[1] = bp[1];
        s[fn].v = __builtin_amdgcn_wmma_f32_16x16x32_bf16(
            false, a.v, false, bb.v, (short)0, s[fn].v, false, false);
      }
    }

    // Online softmax (row = r + 8*half within wave's 16 rows; 16-lane reduce)
    float corr[8];
#pragma unroll
    for (int r = 0; r < 8; ++r) {
      float m = -3.0e38f;
#pragma unroll
      for (int fn = 0; fn < 4; ++fn) {
        s[fn].f[r] *= 0.125f;
        m = fmaxf(m, s[fn].f[r]);
      }
#pragma unroll
      for (int mk = 1; mk < 16; mk <<= 1) m = fmaxf(m, __shfl_xor(m, mk, 32));
      float nm = fmaxf(mst[r], m);
      corr[r] = __expf(mst[r] - nm);
      mst[r] = nm;
      float rsum = 0.f;
#pragma unroll
      for (int fn = 0; fn < 4; ++fn) {
        float p = __expf(s[fn].f[r] - nm);
        s[fn].f[r] = p;
        rsum += p;
      }
#pragma unroll
      for (int mk = 1; mk < 16; mk <<= 1) rsum += __shfl_xor(rsum, mk, 32);
      lst[r] = lst[r] * corr[r] + rsum;
    }
#pragma unroll
    for (int c = 0; c < 4; ++c)
#pragma unroll
      for (int r = 0; r < 8; ++r) o[c].f[r] *= corr[r];

    // Spill P (16x64) to wave-private LDS to reshape C-layout -> A-layout
#pragma unroll
    for (int fn = 0; fn < 4; ++fn)
#pragma unroll
      for (int r = 0; r < 8; ++r)
        Ps[wave * (16 * ASTR) + (r + half * 8) * ASTR + fn * 16 + lr] =
            f2bf(s[fn].f[r]);

    // O += P V   (B-fragment from transposed Vt: contiguous 32B per lane)
#pragma unroll
    for (int kk = 0; kk < 64; kk += 32) {
      AFrag a;
      a.q[0] = *(const u32x4*)(&Ps[wave * (16 * ASTR) + lr * ASTR + kk + half * 16]);
      a.q[1] = *(const u32x4*)(&Ps[wave * (16 * ASTR) + lr * ASTR + kk + half * 16 + 8]);
#pragma unroll
      for (int c = 0; c < 4; ++c) {
        AFrag bb;
        const u32x4* bp = (const u32x4*)(&Vt[(c * 16 + lr) * ASTR + kk + half * 16]);
        bb.q[0] = bp[0];
        bb.q[1] = bp[1];
        o[c].v = __builtin_amdgcn_wmma_f32_16x16x32_bf16(
            false, a.v, false, bb.v, (short)0, o[c].v, false, false);
      }
    }
  }

  // Normalize and store
#pragma unroll
  for (int c = 0; c < 4; ++c)
#pragma unroll
    for (int r = 0; r < 8; ++r) {
      int row = qb + wave * 16 + half * 8 + r;
      int col = h * NDH + c * 16 + lr;
      O[((size_t)b * NT + row) * ND + col] = f2bf(o[c].f[r] / lst[r]);
    }
}

// ---------------------------------------------------------------------------
// LayerNorm -> bf16 (one token per block)
// ---------------------------------------------------------------------------
__global__ __launch_bounds__(256) void ln_kernel(
    const float* __restrict__ x, const float* __restrict__ g,
    const float* __restrict__ bt, unsigned short* __restrict__ out) {
  __shared__ float r1[8], r2[8];
  __shared__ float smean, srstd;
  const int row = blockIdx.x, tid = threadIdx.x;
  const int lane = tid & 31, wave = tid >> 5;
  const float* xr = x + (size_t)row * ND;
  float s = 0.f, s2 = 0.f;
  for (int d = tid; d < ND; d += 256) { float v = xr[d]; s += v; s2 += v * v; }
  for (int mk = 16; mk; mk >>= 1) { s += __shfl_xor(s, mk, 32); s2 += __shfl_xor(s2, mk, 32); }
  if (!lane) { r1[wave] = s; r2[wave] = s2; }
  __syncthreads();
  if (!tid) {
    float S = 0.f, S2 = 0.f;
    for (int w = 0; w < 8; ++w) { S += r1[w]; S2 += r2[w]; }
    float mean = S / (float)ND;
    smean = mean;
    srstd = rsqrtf(S2 / (float)ND - mean * mean + 1e-5f);
  }
  __syncthreads();
  float mean = smean, rstd = srstd;
  for (int d = tid; d < ND; d += 256)
    out[(size_t)row * ND + d] = f2bf((xr[d] - mean) * rstd * g[d] + bt[d]);
}

// ---------------------------------------------------------------------------
// LN2 + router logits + softplus-temp + top-2 sigmoid suppression + softmax
// ---------------------------------------------------------------------------
__global__ __launch_bounds__(256) void ln_router_kernel(
    const float* __restrict__ x, const float* __restrict__ g,
    const float* __restrict__ bt, const float* __restrict__ rw,  // [NE,ND]
    const float* __restrict__ log_temp,
    unsigned short* __restrict__ h2b, float* __restrict__ wout) {
  __shared__ float r1[8], r2[8], slog[8];
  __shared__ float smean, srstd;
  const int row = blockIdx.x, tid = threadIdx.x;
  const int lane = tid & 31, wave = tid >> 5;
  const float* xr = x + (size_t)row * ND;
  float s = 0.f, s2 = 0.f;
  for (int d = tid; d < ND; d += 256) { float v = xr[d]; s += v; s2 += v * v; }
  for (int mk = 16; mk; mk >>= 1) { s += __shfl_xor(s, mk, 32); s2 += __shfl_xor(s2, mk, 32); }
  if (!lane) { r1[wave] = s; r2[wave] = s2; }
  __syncthreads();
  if (!tid) {
    float S = 0.f, S2 = 0.f;
    for (int w = 0; w < 8; ++w) { S += r1[w]; S2 += r2[w]; }
    float mean = S / (float)ND;
    smean = mean;
    srstd = rsqrtf(S2 / (float)ND - mean * mean + 1e-5f);
  }
  __syncthreads();
  const float mean = smean, rstd = srstd;
  for (int d = tid; d < ND; d += 256)
    h2b[(size_t)row * ND + d] = f2bf((xr[d] - mean) * rstd * g[d] + bt[d]);
  // wave `wave` computes logit for expert `wave`
  float acc = 0.f;
  for (int d = lane; d < ND; d += 32) {
    float hv = (xr[d] - mean) * rstd * g[d] + bt[d];
    acc += hv * rw[(size_t)wave * ND + d];
  }
  for (int mk = 16; mk; mk >>= 1) acc += __shfl_xor(acc, mk, 32);
  if (!lane) slog[wave] = acc;
  __syncthreads();
  if (!tid) {
    float temp = log1pf(__expf(log_temp[0])) + 0.1f;
    float lg[NE], m1 = -3.0e38f, m2 = -3.0e38f;
    for (int e = 0; e < NE; ++e) {
      lg[e] = slog[e] / temp;
      if (lg[e] > m1) { m2 = m1; m1 = lg[e]; }
      else if (lg[e] > m2) m2 = lg[e];
    }
    float z[NE], mx = -3.0e38f;
    for (int e = 0; e < NE; ++e) {
      float sup = 1.f / (1.f + __expf(-10.f * (lg[e] - m2)));
      z[e] = lg[e] * sup;
      mx = fmaxf(mx, z[e]);
    }
    float ssum = 0.f;
    for (int e = 0; e < NE; ++e) { z[e] = __expf(z[e] - mx); ssum += z[e]; }
    for (int e = 0; e < NE; ++e) wout[(size_t)row * NE + e] = z[e] / ssum;
  }
}

// ---------------------------------------------------------------------------
// Helpers: f32 -> bf16, and Wd [E,D,DE] -> Wdt [D, E*DE] bf16
// ---------------------------------------------------------------------------
__global__ void cvt_bf16_kernel(const float* __restrict__ in,
                                unsigned short* __restrict__ out, int n) {
  int i = blockIdx.x * 256 + threadIdx.x;
  if (i < n) out[i] = f2bf(in[i]);
}

__global__ void wd_transpose_kernel(const float* __restrict__ Wd,
                                    unsigned short* __restrict__ Wdt) {
  int i = blockIdx.x * 256 + threadIdx.x;
  if (i < NE * ND * NDE) {
    int e = i / (ND * NDE);
    int r = i - e * (ND * NDE);
    int d = r / NDE;
    int f = r - d * NDE;
    Wdt[(size_t)d * NGU + e * NDE + f] = f2bf(Wd[i]);
  }
}

// ---------------------------------------------------------------------------
extern "C" void kernel_launch(void* const* d_in, const int* in_sizes, int n_in,
                              void* d_out, int out_size, void* d_ws,
                              size_t ws_size, hipStream_t stream) {
  (void)in_sizes; (void)n_in; (void)out_size; (void)ws_size;
  const float* x        = (const float*)d_in[0];
  const float* ln1_g    = (const float*)d_in[1];
  const float* ln1_b    = (const float*)d_in[2];
  const float* in_w     = (const float*)d_in[3];
  const float* in_b     = (const float*)d_in[4];
  const float* out_w    = (const float*)d_in[5];
  const float* out_b    = (const float*)d_in[6];
  const float* ln2_g    = (const float*)d_in[7];
  const float* ln2_b    = (const float*)d_in[8];
  const float* router_w = (const float*)d_in[9];
  const float* log_temp = (const float*)d_in[10];
  const float* Wg       = (const float*)d_in[11];
  const float* Wu       = (const float*)d_in[12];
  const float* Wd       = (const float*)d_in[13];

  char* ws = (char*)d_ws;
  size_t cur = 0;
  auto alloc = [&](size_t bytes) -> void* {
    void* p = ws + cur;
    cur += (bytes + 255) & ~(size_t)255;
    return p;
  };
  unsigned short* h1b   = (unsigned short*)alloc((size_t)MTOK * ND * 2);
  unsigned short* wqkvb = (unsigned short*)alloc((size_t)NQKV * ND * 2);
  unsigned short* qkv   = (unsigned short*)alloc((size_t)MTOK * NQKV * 2);
  unsigned short* ob    = (unsigned short*)alloc((size_t)MTOK * ND * 2);
  unsigned short* woutb = (unsigned short*)alloc((size_t)ND * ND * 2);
  float*          xattn = (float*)alloc((size_t)MTOK * ND * 4);
  unsigned short* h2b   = (unsigned short*)alloc((size_t)MTOK * ND * 2);
  float*          wgt   = (float*)alloc((size_t)MTOK * NE * 4);
  unsigned short* wgb   = (unsigned short*)alloc((size_t)NGU * ND * 2);
  unsigned short* wub   = (unsigned short*)alloc((size_t)NGU * ND * 2);
  unsigned short* wdt   = (unsigned short*)alloc((size_t)ND * NGU * 2);
  unsigned short* act2  = (unsigned short*)alloc((size_t)MTOK * NGU * 2);

  // Weight downconversion (once per launch; deterministic)
  {
    int n = NQKV * ND;
    cvt_bf16_kernel<<<(n + 255) / 256, 256, 0, stream>>>(in_w, wqkvb, n);
    n = ND * ND;
    cvt_bf16_kernel<<<(n + 255) / 256, 256, 0, stream>>>(out_w, woutb, n);
    n = NGU * ND;
    cvt_bf16_kernel<<<(n + 255) / 256, 256, 0, stream>>>(Wg, wgb, n);
    cvt_bf16_kernel<<<(n + 255) / 256, 256, 0, stream>>>(Wu, wub, n);
    n = NE * ND * NDE;
    wd_transpose_kernel<<<(n + 255) / 256, 256, 0, stream>>>(Wd, wdt);
  }

  // 1) LN1 -> bf16
  ln_kernel<<<MTOK, 256, 0, stream>>>(x, ln1_g, ln1_b, h1b);

  // 2) QKV projection (bf16 out, +bias)
  gemm_bf16_kernel<true><<<dim3(NQKV / TN, MTOK / TM), 256, 0, stream>>>(
      h1b, wqkvb, in_b, nullptr, qkv, MTOK, NQKV, ND);

  // 3) Flash attention
  attn_kernel<<<dim3(NT / 64, NB * NH), 128, 0, stream>>>(qkv, ob);

  // 4) Out projection + bias + residual x -> xattn (f32)
  gemm_bf16_kernel<false><<<dim3(ND / TN, MTOK / TM), 256, 0, stream>>>(
      ob, woutb, out_b, x, xattn, MTOK, ND, ND);

  // 5) LN2 + router weights
  ln_router_kernel<<<MTOK, 256, 0, stream>>>(xattn, ln2_g, ln2_b, router_w,
                                             log_temp, h2b, wgt);

  // 6) Fused gate/up + SiLU*up*weight -> act2 (bf16)
  gemm_gateup_kernel<<<dim3(NGU / TN, MTOK / TM), 256, 0, stream>>>(
      h2b, wgb, wub, wgt, act2, MTOK, NGU, ND);

  // 7) Down projection + residual xattn -> d_out (f32)
  gemm_bf16_kernel<false><<<dim3(ND / TN, MTOK / TM), 256, 0, stream>>>(
      act2, wdt, nullptr, xattn, (float*)d_out, MTOK, ND, NGU);
}
